// SoMPredictor_25795573579869
// MI455X (gfx1250) — matrix operand order repaired and checked by reading, not compiled, not run on hardware
//
#include <hip/hip_runtime.h>

// ---------------------------------------------------------------------------
// GAT pipeline for MI455X (gfx1250, wave32).
// GEMMs: v_wmma_f32_16x16x32_f16, LDS-staged, bias/ReLU fused in epilogue.
// Edge phase: wave-per-edge gather + L2 atomics (xl/agg fit in 192MB L2).
// ---------------------------------------------------------------------------

typedef __attribute__((ext_vector_type(16))) _Float16 v16h;
typedef __attribute__((ext_vector_type(8)))  float    v8f;

static constexpr int TPB = 256;   // 8 waves of 32

// ---------------- small helpers ----------------

__device__ __forceinline__ float wred32(float v) {
  #pragma unroll
  for (int m = 16; m > 0; m >>= 1) v += __shfl_xor(v, m, 32);
  return v;
}

// order-preserving float<->uint encoding for atomicMax on signed floats
__device__ __forceinline__ unsigned encf(float x) {
  unsigned u = __float_as_uint(x);
  return (u & 0x80000000u) ? ~u : (u | 0x80000000u);
}
__device__ __forceinline__ float decf(unsigned e) {
  return (e & 0x80000000u) ? __uint_as_float(e & 0x7fffffffu)
                           : __uint_as_float(~e);
}

__device__ __forceinline__ void edge_sd(const int* __restrict__ ei, int e,
                                        int E, int& s, int& d) {
  if (e < E) { s = ei[e]; d = ei[E + e]; }      // edge_index[0][e], [1][e]
  else       { s = e - E; d = s; }              // self loop
}

// ---------------- elementwise utility kernels ----------------

__global__ __launch_bounds__(TPB) void k_f2h(const float* __restrict__ in,
                                             _Float16* __restrict__ out, int n) {
  int i = blockIdx.x * TPB + threadIdx.x;
  if (i < n) out[i] = (_Float16)in[i];
}

__global__ __launch_bounds__(TPB) void k_zero(float* __restrict__ p, int n) {
  int i = blockIdx.x * TPB + threadIdx.x;
  if (i < n) p[i] = 0.0f;
}

// ---------------- WMMA GEMM: C[nrows,NOUT] = A[nrows,K] (f16) @ W[K,NOUT] ----
// 64 rows per block, 8 waves; wave w: row-tile w&3, col-tiles (w>>2)*CT..+CT-1.

template<int K, int NOUT, bool RELU>
__global__ __launch_bounds__(TPB)
void k_gemm(const _Float16* __restrict__ A, const float* __restrict__ W,
            const float* __restrict__ bias, float* __restrict__ outF,
            _Float16* __restrict__ outH, int nrows) {
  static_assert(K % 64 == 0 && (NOUT == 64 || NOUT == 128), "shape");
  constexpr int KT = 64;            // K tile
  constexpr int CT = NOUT / 32;     // col-tiles per wave (2 or 4)
  __shared__ _Float16 sA[64 * KT];          // row-major 64 x KT
  __shared__ _Float16 sB[NOUT * KT];        // transposed: [n][k]

  const int tid  = threadIdx.x;
  const int lane = tid & 31;
  const int wave = tid >> 5;
  const int row0 = blockIdx.x * 64;
  const int rt   = wave & 3;                // row tile (16 rows)
  const int c0   = (wave >> 2) * CT;        // first col tile

  v8f acc[CT];
  #pragma unroll
  for (int j = 0; j < CT; ++j)
    #pragma unroll
    for (int t = 0; t < 8; ++t) acc[j][t] = 0.0f;

  for (int kk = 0; kk < K / KT; ++kk) {
    if (kk) __syncthreads();
    // stage A tile (f16 global -> LDS), zero-fill OOB rows
    for (int idx = tid; idx < 64 * (KT / 4); idx += TPB) {
      int r  = idx / (KT / 4);
      int c4 = (idx % (KT / 4)) * 4;
      int gr = row0 + r;
      uint2 v = make_uint2(0u, 0u);
      if (gr < nrows)
        v = *(const uint2*)(A + (size_t)gr * K + kk * KT + c4);
      *(uint2*)(sA + r * KT + c4) = v;
    }
    // stage W tile transposed, fp32 -> f16
    for (int idx = tid; idx < NOUT * KT; idx += TPB) {
      int n = idx / KT, k = idx % KT;
      sB[idx] = (_Float16)W[(size_t)(kk * KT + k) * NOUT + n];
    }
    __syncthreads();

    #pragma unroll
    for (int ks = 0; ks < 2; ++ks) {        // two K=32 WMMA steps per tile
      // A fragment: lane holds row rt*16+(lane&15);
      // elems 0-7: K = ks*32 + hi*8 .. +7 ; elems 8-15: +16   (ISA 7.12.2)
      const int hi = lane >> 4;
      const int ar = rt * 16 + (lane & 15);
      const int ka = ks * 32 + hi * 8;
      union { uint4 u[2]; v16h v; } fa;
      fa.u[0] = *(const uint4*)(sA + ar * KT + ka);
      fa.u[1] = *(const uint4*)(sA + ar * KT + ka + 16);

      #pragma unroll
      for (int j = 0; j < CT; ++j) {
        // B fragment: lane holds column (c0+j)*16+(lane&15);
        // elems 0-7: K = ks*32 + hi*8 .. +7 ; elems 8-15: +16 (same as A)
        const int bc = (c0 + j) * 16 + (lane & 15);
        const int kb = ks * 32 + hi * 8;
        union { uint4 u[2]; v16h v; } fb;
        fb.u[0] = *(const uint4*)(sB + bc * KT + kb);
        fb.u[1] = *(const uint4*)(sB + bc * KT + kb + 16);
        acc[j] = __builtin_amdgcn_wmma_f32_16x16x32_f16(
            false, fa.v, false, fb.v, (short)0, acc[j], false, false);
      }
    }
  }

  // epilogue: D layout — VGPR v: lanes 0-15 M=v, lanes 16-31 M=8+v; N=lane&15
  const int m0 = (lane >> 4) * 8;
  const int nc = lane & 15;
  #pragma unroll
  for (int j = 0; j < CT; ++j) {
    const int col = (c0 + j) * 16 + nc;
    const float bv = bias ? bias[col] : 0.0f;
    #pragma unroll
    for (int v = 0; v < 8; ++v) {
      const int r = row0 + rt * 16 + m0 + v;
      if (r < nrows) {
        float val = acc[j][v] + bv;
        if (RELU) val = fmaxf(val, 0.0f);
        if (outF) outF[(size_t)r * NOUT + col] = val;
        if (outH) outH[(size_t)r * NOUT + col] = (_Float16)val;
      }
    }
  }
}

// ---------------- layernorm (+ReLU) : wave per node, H=128 ----------------

__global__ __launch_bounds__(TPB)
void k_ln_relu(const float* __restrict__ pre, const float* __restrict__ g,
               const float* __restrict__ b, float* __restrict__ h,
               _Float16* __restrict__ hb, int n) {
  int node = blockIdx.x * 8 + (threadIdx.x >> 5);
  if (node >= n) return;
  int lane = threadIdx.x & 31;
  const float* p = pre + (size_t)node * 128;
  float x[4], s = 0.f, s2 = 0.f;
  #pragma unroll
  for (int k = 0; k < 4; ++k) {
    float v = p[lane + 32 * k];
    x[k] = v; s += v; s2 += v * v;
  }
  s = wred32(s); s2 = wred32(s2);
  float mu = s * (1.f / 128.f);
  float rs = rsqrtf(s2 * (1.f / 128.f) - mu * mu + 1e-5f);
  #pragma unroll
  for (int k = 0; k < 4; ++k) {
    int c = lane + 32 * k;
    float v = fmaxf((x[k] - mu) * rs * g[c] + b[c], 0.0f);
    h[(size_t)node * 128 + c]  = v;
    hb[(size_t)node * 128 + c] = (_Float16)v;
  }
}

// GAT finalize: t = h + elu(agg + bias); h = layernorm(t); also f16 copy.
__global__ __launch_bounds__(TPB)
void k_gat_finalize(const float* __restrict__ agg, const float* __restrict__ bias,
                    const float* __restrict__ g, const float* __restrict__ b,
                    float* __restrict__ h, _Float16* __restrict__ hb, int n) {
  int node = blockIdx.x * 8 + (threadIdx.x >> 5);
  if (node >= n) return;
  int lane = threadIdx.x & 31;
  float x[4], s = 0.f, s2 = 0.f;
  #pragma unroll
  for (int k = 0; k < 4; ++k) {
    int c = lane + 32 * k;
    float a = agg[(size_t)node * 128 + c] + bias[c];
    a = a > 0.f ? a : expm1f(a);                 // ELU
    float v = h[(size_t)node * 128 + c] + a;     // residual
    x[k] = v; s += v; s2 += v * v;
  }
  s = wred32(s); s2 = wred32(s2);
  float mu = s * (1.f / 128.f);
  float rs = rsqrtf(s2 * (1.f / 128.f) - mu * mu + 1e-5f);
  #pragma unroll
  for (int k = 0; k < 4; ++k) {
    int c = lane + 32 * k;
    float v = (x[k] - mu) * rs * g[c] + b[c];
    h[(size_t)node * 128 + c]  = v;
    hb[(size_t)node * 128 + c] = (_Float16)v;
  }
}

// ---------------- attention ----------------

// per-node head dots: a_s[n,h] = sum_c xl[n,h*32+c]*as[h,c]  (wave per node)
__global__ __launch_bounds__(TPB)
void k_att_node(const float* __restrict__ xl, const float* __restrict__ as,
                const float* __restrict__ ad, float* __restrict__ a_s,
                float* __restrict__ a_d, int n) {
  int node = blockIdx.x * 8 + (threadIdx.x >> 5);
  if (node >= n) return;
  int lane = threadIdx.x & 31;
  #pragma unroll
  for (int h = 0; h < 4; ++h) {
    float v  = xl[(size_t)node * 128 + h * 32 + lane];
    float ps = wred32(v * as[h * 32 + lane]);
    float pd = wred32(v * ad[h * 32 + lane]);
    if (lane == 0) { a_s[node * 4 + h] = ps; a_d[node * 4 + h] = pd; }
  }
}

// pass 1: segment max over dst (encoded-uint atomicMax); thread per edge
__global__ __launch_bounds__(TPB)
void k_edge_max(const int* __restrict__ ei, const float* __restrict__ a_s,
                const float* __restrict__ a_d, unsigned* __restrict__ m,
                const float* __restrict__ xl, int E, int N) {
  int e = blockIdx.x * TPB + threadIdx.x;
  if (e >= E + N) return;
  int s, d; edge_sd(ei, e, E, s, d);
  __builtin_prefetch(xl + (size_t)s * 128, 0, 1);   // warm L2 for pass 3
  #pragma unroll
  for (int h = 0; h < 4; ++h) {
    float v = a_s[s * 4 + h] + a_d[d * 4 + h];
    v = v > 0.f ? v : 0.2f * v;                     // leaky_relu(0.2)
    atomicMax(&m[d * 4 + h], encf(v));
  }
}

// pass 2: denom[dst] += exp(e - m[dst]); thread per edge
__global__ __launch_bounds__(TPB)
void k_edge_expsum(const int* __restrict__ ei, const float* __restrict__ a_s,
                   const float* __restrict__ a_d, const unsigned* __restrict__ m,
                   float* __restrict__ denom, int E, int N) {
  int e = blockIdx.x * TPB + threadIdx.x;
  if (e >= E + N) return;
  int s, d; edge_sd(ei, e, E, s, d);
  #pragma unroll
  for (int h = 0; h < 4; ++h) {
    float v = a_s[s * 4 + h] + a_d[d * 4 + h];
    v = v > 0.f ? v : 0.2f * v;
    atomicAdd(&denom[d * 4 + h], __expf(v - decf(m[d * 4 + h])));
  }
}

// pass 3: agg[dst] += alpha * xl[src]; wave per edge (lane = channel%32)
__global__ __launch_bounds__(TPB)
void k_edge_aggr(const int* __restrict__ ei, const float* __restrict__ a_s,
                 const float* __restrict__ a_d, const unsigned* __restrict__ m,
                 const float* __restrict__ denom, const float* __restrict__ xl,
                 float* __restrict__ agg, int E, int N) {
  int e = blockIdx.x * 8 + (threadIdx.x >> 5);
  if (e >= E + N) return;
  int lane = threadIdx.x & 31;
  int s, d; edge_sd(ei, e, E, s, d);
  __builtin_prefetch(xl + (size_t)s * 128 + lane * 4, 0, 1);
  float al[4];
  #pragma unroll
  for (int h = 0; h < 4; ++h) {
    float v = a_s[s * 4 + h] + a_d[d * 4 + h];
    v = v > 0.f ? v : 0.2f * v;
    al[h] = __expf(v - decf(m[d * 4 + h])) / denom[d * 4 + h];
  }
  #pragma unroll
  for (int h = 0; h < 4; ++h) {
    int c = h * 32 + lane;
    atomicAdd(&agg[(size_t)d * 128 + c], xl[(size_t)s * 128 + c] * al[h]);
  }
}

// ---------------- pooling / classifier tails ----------------

__global__ __launch_bounds__(TPB)
void k_pool_accum(const float* __restrict__ h, const int* __restrict__ batch,
                  float* __restrict__ sums, float* __restrict__ cnt, int n) {
  int node = blockIdx.x * 8 + (threadIdx.x >> 5);
  if (node >= n) return;
  int lane = threadIdx.x & 31;
  int g = batch[node];
  #pragma unroll
  for (int k = 0; k < 4; ++k) {
    int c = lane + 32 * k;
    atomicAdd(&sums[(size_t)g * 128 + c], h[(size_t)node * 128 + c]);
  }
  if (lane == 0) atomicAdd(&cnt[g], 1.0f);
}

__global__ __launch_bounds__(TPB)
void k_comb(const float* __restrict__ h, const float* __restrict__ sums,
            const float* __restrict__ cnt, const int* __restrict__ batch,
            _Float16* __restrict__ comb, int n) {
  int node = blockIdx.x * 8 + (threadIdx.x >> 5);
  if (node >= n) return;
  int lane = threadIdx.x & 31;
  int g = batch[node];
  float inv = 1.0f / fmaxf(cnt[g], 1.0f);
  #pragma unroll
  for (int k = 0; k < 4; ++k) {
    int c = lane + 32 * k;
    comb[(size_t)node * 256 + c]       = (_Float16)h[(size_t)node * 128 + c];
    comb[(size_t)node * 256 + 128 + c] = (_Float16)(sums[(size_t)g * 128 + c] * inv);
  }
}

// out[n] = z2[n,:] . W3 + b3   (wave per node, 64-wide dot)
__global__ __launch_bounds__(TPB)
void k_final(const float* __restrict__ z2, const float* __restrict__ W3,
             const float* __restrict__ b3, float* __restrict__ out, int n) {
  int node = blockIdx.x * 8 + (threadIdx.x >> 5);
  if (node >= n) return;
  int lane = threadIdx.x & 31;
  float v = z2[(size_t)node * 64 + lane] * W3[lane]
          + z2[(size_t)node * 64 + 32 + lane] * W3[32 + lane];
  v = wred32(v);
  if (lane == 0) out[node] = v + b3[0];
}

// ---------------------------------------------------------------------------

extern "C" void kernel_launch(void* const* d_in, const int* in_sizes, int n_in,
                              void* d_out, int out_size, void* d_ws, size_t ws_size,
                              hipStream_t stream) {
  const int N = in_sizes[0] / 64;     // 100000
  const int E = in_sizes[1] / 2;      // 1600000
  const int EE = E + N;               // with self loops
  const int G = 2000;                 // graphs (reference constant)

  const float* x      = (const float*)d_in[0];
  const int*   ei     = (const int*)  d_in[1];
  const int*   batch  = (const int*)  d_in[2];
  const float* in_W   = (const float*)d_in[3];
  const float* in_b   = (const float*)d_in[4];
  const float* ln0_g  = (const float*)d_in[5];
  const float* ln0_b  = (const float*)d_in[6];
  const float* cls_W1 = (const float*)d_in[7];
  const float* cls_b1 = (const float*)d_in[8];
  const float* cls_W2 = (const float*)d_in[9];
  const float* cls_b2 = (const float*)d_in[10];
  const float* cls_W3 = (const float*)d_in[11];
  const float* cls_b3 = (const float*)d_in[12];

  // workspace carve-out (256B aligned)
  char* w = (char*)d_ws;
  auto alloc = [&](size_t bytes) -> void* {
    void* p = (void*)w;
    w += (bytes + 255) & ~(size_t)255;
    return p;
  };
  _Float16* xh   = (_Float16*)alloc((size_t)N * 64 * 2);
  float*    h    = (float*)   alloc((size_t)N * 128 * 4);
  _Float16* hb   = (_Float16*)alloc((size_t)N * 128 * 2);
  float*    f0   = (float*)   alloc((size_t)N * 128 * 4);   // pre-LN / xl
  float*    agg  = (float*)   alloc((size_t)N * 128 * 4);
  float*    a_s  = (float*)   alloc((size_t)N * 4 * 4);
  float*    a_d  = (float*)   alloc((size_t)N * 4 * 4);
  float*    md   = (float*)   alloc((size_t)N * 8 * 4);     // m (uint) + denom
  unsigned* m    = (unsigned*)md;
  float*    den  = md + (size_t)N * 4;
  float*    sums = (float*)   alloc((size_t)G * 129 * 4);   // sums[G,128] + cnt[G]
  float*    cnt  = sums + (size_t)G * 128;
  _Float16* comb = (_Float16*)alloc((size_t)N * 256 * 2);
  _Float16* z1h  = (_Float16*)alloc((size_t)N * 128 * 2);
  float*    z2   = (float*)   alloc((size_t)N * 64 * 4);

  const int gemmBlocks = (N + 63) / 64;
  const int nodeBlocks = (N + 7) / 8;

  // 1. input projection: h = relu(LN(x @ in_W + in_b))
  k_f2h<<<(N * 64 + TPB - 1) / TPB, TPB, 0, stream>>>(x, xh, N * 64);
  k_gemm<64, 128, false><<<gemmBlocks, TPB, 0, stream>>>(xh, in_W, in_b, f0, ((_Float16*)nullptr), N);
  k_ln_relu<<<nodeBlocks, TPB, 0, stream>>>(f0, ln0_g, ln0_b, h, hb, N);

  // 2. three GAT layers
  for (int L = 0; L < 3; ++L) {
    const float* gW  = (const float*)d_in[13 + 6 * L + 0];
    const float* gas = (const float*)d_in[13 + 6 * L + 1];
    const float* gad = (const float*)d_in[13 + 6 * L + 2];
    const float* gb  = (const float*)d_in[13 + 6 * L + 3];
    const float* lg  = (const float*)d_in[13 + 6 * L + 4];
    const float* lb  = (const float*)d_in[13 + 6 * L + 5];

    k_gemm<128, 128, false><<<gemmBlocks, TPB, 0, stream>>>(hb, gW, (const float*)nullptr, f0, (_Float16*)nullptr, N);
    k_zero<<<(N * 128 + TPB - 1) / TPB, TPB, 0, stream>>>(agg, N * 128);
    k_zero<<<(N * 8 + TPB - 1) / TPB, TPB, 0, stream>>>(md, N * 8);  // m=enc floor, denom=0
    k_att_node<<<nodeBlocks, TPB, 0, stream>>>(f0, gas, gad, a_s, a_d, N);
    k_edge_max<<<(EE + TPB - 1) / TPB, TPB, 0, stream>>>(ei, a_s, a_d, m, f0, E, N);
    k_edge_expsum<<<(EE + TPB - 1) / TPB, TPB, 0, stream>>>(ei, a_s, a_d, m, den, E, N);
    k_edge_aggr<<<(EE + 7) / 8, TPB, 0, stream>>>(ei, a_s, a_d, m, den, f0, agg, E, N);
    k_gat_finalize<<<nodeBlocks, TPB, 0, stream>>>(agg, gb, lg, lb, h, hb, N);
  }

  // 3. global mean pool + concat
  k_zero<<<(G * 129 + TPB - 1) / TPB, TPB, 0, stream>>>(sums, G * 129);
  k_pool_accum<<<nodeBlocks, TPB, 0, stream>>>(h, batch, sums, cnt, N);
  k_comb<<<nodeBlocks, TPB, 0, stream>>>(h, sums, cnt, batch, comb, N);

  // 4. classifier
  k_gemm<256, 128, true><<<gemmBlocks, TPB, 0, stream>>>(comb, cls_W1, cls_b1, (float*)nullptr, z1h, N);
  k_gemm<128, 64, true><<<gemmBlocks, TPB, 0, stream>>>(z1h, cls_W2, cls_b2, z2, (_Float16*)nullptr, N);
  k_final<<<nodeBlocks, TPB, 0, stream>>>(z2, cls_W3, cls_b3, (float*)d_out, N);
}
